// image_prj_39926015984314
// MI455X (gfx1250) — matrix operand order repaired and checked by reading, not compiled, not run on hardware
//
#include <hip/hip_runtime.h>
#include <math.h>

#define NMAT 2048
#define BM 128
#define BN 128
#define BK 32
#define LSTR 40              // LDS row stride (bf16 elements), 80B -> 16B aligned
#define PDIM 2198
#define PAD_OFF 75
#define NANG 32

typedef __bf16        bf16x16 __attribute__((ext_vector_type(16)));
typedef float         f32x8   __attribute__((ext_vector_type(8)));
typedef unsigned int  u32x4   __attribute__((ext_vector_type(4)));

union FragU { bf16x16 v; u32x4 u[2]; };

__device__ __forceinline__ unsigned short f2bf(float f) {
  unsigned int x = __float_as_uint(f);
  x += 0x7FFFu + ((x >> 16) & 1u);          // round-to-nearest-even
  return (unsigned short)(x >> 16);
}

// Async global->LDS copy of 32 contiguous bytes (2 x B128). The instruction's
// IOFFSET is added to BOTH the LDS and global addresses (ISA 08 §4.4), so one
// address pair covers both halves. Tracked by ASYNCcnt.
__device__ __forceinline__ void async_cp32(unsigned ldsOff, const void* gaddr) {
  asm volatile("global_load_async_to_lds_b128 %0, %1, off"
               :: "v"(ldsOff), "v"(gaddr) : "memory");
  asm volatile("global_load_async_to_lds_b128 %0, %1, off offset:16"
               :: "v"(ldsOff), "v"(gaddr) : "memory");
}

__device__ __forceinline__ void wait_async0() {
  asm volatile("s_wait_asynccnt 0x0" ::: "memory");
}

// ---------------------------------------------------------------------------
// Build DCT-II (ortho) matrix in bf16: C[k][n] = s_k * cos(pi*(n+0.5)*k/N).
// Exact phase reduction: (2n+1)*k mod 4N keeps cosf argument < 2*pi.
// ---------------------------------------------------------------------------
__global__ void k_build_c(unsigned short* __restrict__ Cbf) {
  int idx = blockIdx.x * blockDim.x + threadIdx.x;   // k*NMAT + n
  int k = idx >> 11;
  int n = idx & (NMAT - 1);
  int m = ((2 * n + 1) * k) & (4 * NMAT - 1);        // mod 8192, exact
  float ang = (float)m * (3.14159265358979323846f / (2.0f * NMAT));
  float s = (k == 0) ? 0.0220970869120796f /*sqrt(1/2048)*/
                     : 0.03125f            /*sqrt(2/2048)*/;
  Cbf[idx] = f2bf(s * cosf(ang));
}

// Xt[n][k] = bf16(image[k][n])  (pre-transposed B operand for GEMM1)
__global__ void k_xt(const float* __restrict__ img, unsigned short* __restrict__ Xt) {
  int idx = blockIdx.x * blockDim.x + threadIdx.x;   // n*NMAT + k
  int n = idx >> 11;
  int k = idx & (NMAT - 1);
  Xt[idx] = f2bf(img[k * NMAT + n]);
}

// ---------------------------------------------------------------------------
// bf16 GEMM, f32 accumulate:  D[M,N] = A[M,K] * Bt[N,K]^T
// Block: 256 threads (8 waves), 128x128 tile, K-step 32.
// Double-buffered LDS filled by GLOBAL_LOAD_ASYNC_TO_LDS_B128 (ASYNCcnt),
// each wave runs a 4x2 grid of v_wmma_f32_16x16x32_bf16.
// ---------------------------------------------------------------------------
template <bool OUT_BF16>
__global__ void k_gemm(const unsigned short* __restrict__ A,
                       const unsigned short* __restrict__ Bt,
                       void* __restrict__ D) {
  __shared__ unsigned short sA[2][BM * LSTR];
  __shared__ unsigned short sB[2][BN * LSTR];

  const int tid  = threadIdx.x;
  const int lane = tid & 31;
  const int wave = tid >> 5;
  const int wm   = wave & 1;          // 2 waves along M (64 rows each)
  const int wn   = wave >> 1;         // 4 waves along N (32 cols each)
  const int half = lane >> 4;
  const int l16  = lane & 15;

  const int rowBase = blockIdx.y * BM;
  const int colBase = blockIdx.x * BN;

  const int ldRow = tid >> 1;         // 0..127
  const int ldSeg = (tid & 1) * 16;   // 0 or 16 (elements)

  // Per-thread LDS byte offsets for the async copy destinations
  // (low 32 bits of a generic shared pointer == LDS byte address).
  unsigned aOff[2], bOff[2];
#pragma unroll
  for (int b = 0; b < 2; ++b) {
    aOff[b] = (unsigned)(uintptr_t)(&sA[b][ldRow * LSTR + ldSeg]);
    bOff[b] = (unsigned)(uintptr_t)(&sB[b][ldRow * LSTR + ldSeg]);
  }
  const unsigned short* gaBase = A  + (size_t)(rowBase + ldRow) * NMAT + ldSeg;
  const unsigned short* gbBase = Bt + (size_t)(colBase + ldRow) * NMAT + ldSeg;

  f32x8 acc[4][2];
#pragma unroll
  for (int mt = 0; mt < 4; ++mt)
#pragma unroll
    for (int nt = 0; nt < 2; ++nt)
      acc[mt][nt] = {};

  // Prologue: fill buffer 0.
  async_cp32(aOff[0], gaBase);
  async_cp32(bOff[0], gbBase);
  wait_async0();
  __syncthreads();

  int buf = 0;
  for (int k0 = 0; k0 < NMAT; k0 += BK) {
    // Prefetch next K-slab into the other buffer (no wait before compute).
    if (k0 + BK < NMAT) {
      async_cp32(aOff[buf ^ 1], gaBase + k0 + BK);
      async_cp32(bOff[buf ^ 1], gbBase + k0 + BK);
    }

    const unsigned short* cA = sA[buf];
    const unsigned short* cB = sB[buf];

    // B fragments: lane-half selects K 0..15 / 16..31 (contiguous 32B)
    FragU b[2];
#pragma unroll
    for (int nt = 0; nt < 2; ++nt) {
      int brow = wn * 32 + nt * 16 + l16;
      b[nt].u[0] = *(const u32x4*)(cB + brow * LSTR + half * 16);
      b[nt].u[1] = *(const u32x4*)(cB + brow * LSTR + half * 16 + 8);
    }
#pragma unroll
    for (int mt = 0; mt < 4; ++mt) {
      // A fragment: lane-half selects K 0..15 / 16..31 (contiguous 32B)
      FragU a;
      int arow = wm * 64 + mt * 16 + l16;
      a.u[0] = *(const u32x4*)(cA + arow * LSTR + half * 16);
      a.u[1] = *(const u32x4*)(cA + arow * LSTR + half * 16 + 8);
#pragma unroll
      for (int nt = 0; nt < 2; ++nt) {
        acc[mt][nt] = __builtin_amdgcn_wmma_f32_16x16x32_bf16(
            false, a.v, false, b[nt].v, (short)0, acc[mt][nt], false, false);
      }
    }

    // Prefetch must land and every wave must be done reading `buf` before
    // the next iteration computes / overwrites.
    wait_async0();
    __syncthreads();
    buf ^= 1;
  }

  // Epilogue: D layout — VGPR v holds row (v + 8*half), lane%16 is column.
#pragma unroll
  for (int mt = 0; mt < 4; ++mt)
#pragma unroll
    for (int nt = 0; nt < 2; ++nt)
#pragma unroll
      for (int v = 0; v < 8; ++v) {
        int gr = rowBase + wm * 64 + mt * 16 + half * 8 + v;
        int gc = colBase + wn * 32 + nt * 16 + l16;
        if (OUT_BF16)
          ((unsigned short*)D)[(size_t)gr * NMAT + gc] = f2bf(acc[mt][nt][v]);
        else
          ((float*)D)[(size_t)gr * NMAT + gc] = acc[mt][nt][v];
      }
}

// ---------------------------------------------------------------------------
// Rotate (WRAP bilinear) + sum over rows, 32 angles. One thread per
// (angle, output column); incremental coordinate update, no per-step trig.
// padded[y][x] = d[y-75][x-75] inside the 2048x2048 window, else 0.
// ---------------------------------------------------------------------------
__device__ __forceinline__ float samp(const float* __restrict__ d, int y, int x) {
  y -= PAD_OFF; x -= PAD_OFF;
  if ((unsigned)y < (unsigned)NMAT && (unsigned)x < (unsigned)NMAT)
    return d[y * NMAT + x];
  return 0.0f;
}

__global__ void k_rot(const float* __restrict__ d, float* __restrict__ lines) {
  int t = blockIdx.x * blockDim.x + threadIdx.x;
  if (t >= NANG * PDIM) return;
  int a = t / PDIM;
  int x = t - a * PDIM;

  float theta = 3.14159265358979323846f * (float)a / 31.0f;
  float c = cosf(theta), s = sinf(theta);
  const float cx = (float)(PDIM / 2);

  float xin = c * (float)x - cx * (c + s - 1.0f);    // value at y = 0
  float yin = -s * (float)x - cx * (c - s - 1.0f);
  float accum = 0.0f;

  for (int y = 0; y < PDIM; ++y) {
    float xf = floorf(xin), yf = floorf(yin);
    float wx = xin - xf,   wy = yin - yf;
    int x0 = (int)xf % PDIM; if (x0 < 0) x0 += PDIM;
    int y0 = (int)yf % PDIM; if (y0 < 0) y0 += PDIM;
    int x1 = x0 + 1; if (x1 == PDIM) x1 = 0;
    int y1 = y0 + 1; if (y1 == PDIM) y1 = 0;
    float v00 = samp(d, y0, x0);
    float v01 = samp(d, y0, x1);
    float v10 = samp(d, y1, x0);
    float v11 = samp(d, y1, x1);
    accum += (1.0f - wy) * ((1.0f - wx) * v00 + wx * v01)
           +         wy  * ((1.0f - wx) * v10 + wx * v11);
    xin += s;
    yin += c;
  }
  lines[t] = accum;
}

// ---------------------------------------------------------------------------
// Global max over lines, then out[x][a] = lines[a][x] / max. Single block.
// ---------------------------------------------------------------------------
__global__ void k_norm(const float* __restrict__ lines, float* __restrict__ out) {
  __shared__ float red[256];
  const int total = NANG * PDIM;
  float m = -INFINITY;
  for (int i = threadIdx.x; i < total; i += blockDim.x)
    m = fmaxf(m, lines[i]);
  red[threadIdx.x] = m;
  __syncthreads();
  for (int s = 128; s > 0; s >>= 1) {
    if ((int)threadIdx.x < s)
      red[threadIdx.x] = fmaxf(red[threadIdx.x], red[threadIdx.x + s]);
    __syncthreads();
  }
  float inv = 1.0f / red[0];
  for (int i = threadIdx.x; i < total; i += blockDim.x) {
    int a = i / PDIM;
    int x = i - a * PDIM;
    out[x * NANG + a] = lines[i] * inv;
  }
}

// ---------------------------------------------------------------------------
extern "C" void kernel_launch(void* const* d_in, const int* in_sizes, int n_in,
                              void* d_out, int out_size, void* d_ws, size_t ws_size,
                              hipStream_t stream) {
  (void)in_sizes; (void)n_in; (void)out_size; (void)ws_size;
  const float* image = (const float*)d_in[0];

  // Workspace carve-up (~40.3 MiB total)
  unsigned short* Cbf  = (unsigned short*)d_ws;          // 2048^2 bf16
  unsigned short* Xt   = Cbf + (size_t)NMAT * NMAT;      // 2048^2 bf16
  unsigned short* T1   = Xt  + (size_t)NMAT * NMAT;      // 2048^2 bf16
  float*          dmat = (float*)(T1 + (size_t)NMAT * NMAT); // 2048^2 f32
  float*          lines = dmat + (size_t)NMAT * NMAT;    // 32*2198 f32

  const int elems = NMAT * NMAT;
  k_build_c<<<elems / 256, 256, 0, stream>>>(Cbf);
  k_xt<<<elems / 256, 256, 0, stream>>>(image, Xt);

  dim3 gg(NMAT / BN, NMAT / BM);
  // T1 = C @ X           (A = C, Bt = X^T)
  k_gemm<true><<<gg, 256, 0, stream>>>(Cbf, Xt, (void*)T1);
  // d  = T1 @ C^T        (A = T1, Bt = C)
  k_gemm<false><<<gg, 256, 0, stream>>>(T1, Cbf, (void*)dmat);

  int rt = NANG * PDIM;
  k_rot<<<(rt + 255) / 256, 256, 0, stream>>>(dmat, lines);
  k_norm<<<1, 256, 0, stream>>>(lines, (float*)d_out);
}
